// MultiHeadAttention_65266323030553
// MI455X (gfx1250) — compile-verified, matrix operand order checked by reading, not run on hardware
//
#include <hip/hip_runtime.h>

#define DMODEL 1024
#define NHEADS 16
#define DHEAD  64
#define BATCH  4
#define TSEQ   2048
#define MTOK   (BATCH * TSEQ)   // 8192
#define KBLK   32

typedef __bf16 bf16;
typedef __attribute__((ext_vector_type(16))) __bf16 bf16x16;
typedef __attribute__((ext_vector_type(8)))  __bf16 bf16x8;
typedef __attribute__((ext_vector_type(4)))  __bf16 bf16x4;
typedef __attribute__((ext_vector_type(8)))  float  f32x8;
typedef __attribute__((ext_vector_type(4)))  float  f32x4;
typedef __attribute__((ext_vector_type(4)))  unsigned int u32x4;
typedef __attribute__((ext_vector_type(8)))  int    i32x8;
typedef __attribute__((ext_vector_type(4)))  int    i32x4;

// ---------------------------------------------------------------------------
// WMMA wrapper: D = A(16x32 bf16) x B(32x16 bf16) + C(16x16 f32)
// ---------------------------------------------------------------------------
__device__ __forceinline__ f32x8 wmma_bf16(bf16x16 a, bf16x16 b, f32x8 c) {
  return __builtin_amdgcn_wmma_f32_16x16x32_bf16(false, a, false, b,
                                                 (short)0, c, false, false);
}

// ---------------------------------------------------------------------------
// 16x32 bf16 operand tile in ISA A-layout (same pattern for NT-GEMM B operand)
// ---------------------------------------------------------------------------
__device__ __forceinline__ bf16x16 ld_tile(const bf16* __restrict__ base, int ld) {
  int lane = threadIdx.x & 31;
  const bf16* p = base + (size_t)(lane & 15) * ld + ((lane >> 4) << 3);
  bf16x8 lo = *(const bf16x8*)p;
  bf16x8 hi = *(const bf16x8*)(p + 16);
  bf16x16 r;
#pragma unroll
  for (int i = 0; i < 8; ++i) { r[i] = lo[i]; r[i + 8] = hi[i]; }
  return r;
}

// ---------------------------------------------------------------------------
// VALU-only 16-lane butterfly reductions via DPP16 row_xmask (no LDS traffic)
// ---------------------------------------------------------------------------
#if __has_builtin(__builtin_amdgcn_update_dpp)
#define DPP_XORF(x, m)                                                         \
  __builtin_bit_cast(float, __builtin_amdgcn_update_dpp(                       \
      __builtin_bit_cast(int, x), __builtin_bit_cast(int, x),                  \
      0x160 | (m), 0xf, 0xf, false))
__device__ __forceinline__ float red_max16(float x) {
  x = fmaxf(x, DPP_XORF(x, 1));
  x = fmaxf(x, DPP_XORF(x, 2));
  x = fmaxf(x, DPP_XORF(x, 4));
  x = fmaxf(x, DPP_XORF(x, 8));
  return x;
}
__device__ __forceinline__ float red_sum16(float x) {
  x += DPP_XORF(x, 1);
  x += DPP_XORF(x, 2);
  x += DPP_XORF(x, 4);
  x += DPP_XORF(x, 8);
  return x;
}
#else
__device__ __forceinline__ float red_max16(float x) {
  for (int d = 8; d >= 1; d >>= 1) x = fmaxf(x, __shfl_xor(x, d, 32));
  return x;
}
__device__ __forceinline__ float red_sum16(float x) {
  for (int d = 8; d >= 1; d >>= 1) x += __shfl_xor(x, d, 32);
  return x;
}
#endif

__device__ __forceinline__ float fast_exp2(float x) {
#if __has_builtin(__builtin_amdgcn_exp2f)
  return __builtin_amdgcn_exp2f(x);   // raw v_exp_f32; arg <= 0 here
#else
  return exp2f(x);
#endif
}

// ---------------------------------------------------------------------------
// Tensor Data Mover: async 2D tile load global -> LDS (row-major in LDS).
// All arguments must be wave-uniform. tile_d0 = contiguous extent (elements),
// tile_d1 = rows, stride0 = row stride in elements, data_size = 2 bytes.
// This toolchain exposes the 6-arg builtin:
//   (u32x4 g0, i32x8 g1, i32x4 g2, i32x4 g3, i32x8 g4, i32 cpol)
// ---------------------------------------------------------------------------
__device__ __forceinline__ void tdm_load_2d(const bf16* gaddr, bf16* ldsp,
                                            unsigned tensor_d0, unsigned tensor_d1,
                                            unsigned tile_d0, unsigned tile_d1,
                                            unsigned stride0) {
#if __has_builtin(__builtin_amdgcn_tensor_load_to_lds)
  unsigned long long ga = (unsigned long long)gaddr;
  unsigned lds_off = (unsigned)(uintptr_t)ldsp;     // low 32 bits = LDS offset
  u32x4 g0;
  g0[0] = 1u;                                        // count=1, user descriptor
  g0[1] = lds_off;                                   // D#.lds_addr
  g0[2] = (unsigned)ga;                              // D#.global_addr[31:0]
  g0[3] = (unsigned)((ga >> 32) & 0x01FFFFFFu) | 0x80000000u;  // ga[56:32] | type=2
  i32x8 g1;
  g1[0] = 0x00010000;                                // workgroup_mask=0, data_size=1 (2B)
  g1[1] = (int)(tensor_d0 << 16);                    // tensor_dim0 lo16
  g1[2] = (int)((tensor_d0 >> 16) | (tensor_d1 << 16));
  g1[3] = (int)((tensor_d1 >> 16) | (tile_d0 << 16));
  g1[4] = (int)tile_d1;                              // tile_dim1; tile_dim2=0
  g1[5] = (int)stride0;                              // tensor_dim0_stride[31:0]
  g1[6] = 0;
  g1[7] = 0;
  i32x4 z4 = {0, 0, 0, 0};
  i32x8 z8 = {0, 0, 0, 0, 0, 0, 0, 0};
  __builtin_amdgcn_tensor_load_to_lds(g0, g1, z4, z4, z8, 0);
#else
  int lane = threadIdx.x & 31;
  unsigned cpr = tile_d0 >> 3;                       // 16B chunks per row
  for (unsigned c = lane; c < cpr * tile_d1; c += 32) {
    unsigned row = c / cpr, col = (c % cpr) << 3;
    *(bf16x8*)(ldsp + (size_t)row * tile_d0 + col) =
        *(const bf16x8*)(gaddr + (size_t)row * stride0 + col);
  }
#endif
}

__device__ __forceinline__ void wait_tensorcnt_2() {
#if __has_builtin(__builtin_amdgcn_s_wait_tensorcnt)
  __builtin_amdgcn_s_wait_tensorcnt(2);
#endif
}
__device__ __forceinline__ void wait_tensorcnt_0() {
#if __has_builtin(__builtin_amdgcn_s_wait_tensorcnt)
  __builtin_amdgcn_s_wait_tensorcnt(0);
#endif
}

// ---------------------------------------------------------------------------
// fp32 -> bf16 bulk convert with optional scale folding
// ---------------------------------------------------------------------------
__global__ void cvt_f32_bf16(const float* __restrict__ in, bf16* __restrict__ out,
                             int n4, float scale) {
  int i = blockIdx.x * blockDim.x + threadIdx.x;
  if (i < n4) {
    f32x4 v = *(const f32x4*)(in + (size_t)i * 4);
    bf16x4 o;
#pragma unroll
    for (int j = 0; j < 4; ++j) o[j] = (bf16)(v[j] * scale);
    *(bf16x4*)(out + (size_t)i * 4) = o;
  }
}

// ---------------------------------------------------------------------------
// NT GEMM: C[m,n] = sum_k A[m,k]*W[n,k].  One wave: 16(M) x 64(N) tile.
// MODE 0: bf16 out [b,h,t,d]; MODE 1: bf16 out [b,h,d,t]; MODE 2: f32 row-major.
// ---------------------------------------------------------------------------
template <int MODE>
__global__ void gemm_nt_kernel(const bf16* __restrict__ A, const bf16* __restrict__ W,
                               void* __restrict__ out) {
  int wid = threadIdx.x >> 5;
  int mt  = blockIdx.x * 4 + wid;
  int nt  = blockIdx.y;
  const bf16* Ab = A + (size_t)mt * 16 * DMODEL;
  const bf16* Wb = W + (size_t)nt * 64 * DMODEL;

  f32x8 c0 = {}, c1 = {}, c2 = {}, c3 = {};
  for (int k0 = 0; k0 < DMODEL; k0 += 32) {
    __builtin_prefetch(Ab + k0 + 128, 0, 1);
    __builtin_prefetch(Wb + k0 + 128, 0, 1);
    bf16x16 a  = ld_tile(Ab + k0, DMODEL);
    bf16x16 w0 = ld_tile(Wb + 0  * DMODEL + k0, DMODEL);
    bf16x16 w1 = ld_tile(Wb + 16 * DMODEL + k0, DMODEL);
    bf16x16 w2 = ld_tile(Wb + 32 * DMODEL + k0, DMODEL);
    bf16x16 w3 = ld_tile(Wb + 48 * DMODEL + k0, DMODEL);
    c0 = wmma_bf16(a, w0, c0);
    c1 = wmma_bf16(a, w1, c1);
    c2 = wmma_bf16(a, w2, c2);
    c3 = wmma_bf16(a, w3, c3);
  }

  int lane = threadIdx.x & 31;
  int half = lane >> 4, cn = lane & 15;
#pragma unroll
  for (int i = 0; i < 8; ++i) {
    int m = mt * 16 + i + 8 * half;
    float vv[4] = { c0[i], c1[i], c2[i], c3[i] };
#pragma unroll
    for (int j = 0; j < 4; ++j) {
      int n = nt * 64 + j * 16 + cn;
      if (MODE == 2) {
        ((float*)out)[(size_t)m * DMODEL + n] = vv[j];
      } else {
        int b = m >> 11, t = m & (TSEQ - 1);
        int h = n >> 6,  d = n & (DHEAD - 1);
        size_t o = (MODE == 0)
          ? (((size_t)(b * NHEADS + h) * TSEQ + t) * DHEAD + d)
          : (((size_t)(b * NHEADS + h) * DHEAD + d) * TSEQ + t);
        ((bf16*)out)[o] = (bf16)vv[j];
      }
    }
  }
}

// ---------------------------------------------------------------------------
// Causal flash attention. 4 waves/block, 64 query rows/block; shared 32-key
// K/V blocks double-buffered in LDS via TDM. Scale pre-folded into Q.
// Q,K: [B*H,T,64] bf16.  Vt: [B*H,64,T] bf16.  O: [B,T,1024] bf16.
// ---------------------------------------------------------------------------
__global__ void flash_attn_kernel(const bf16* __restrict__ Q, const bf16* __restrict__ K,
                                  const bf16* __restrict__ Vt, bf16* __restrict__ O) {
  __shared__ __align__(16) bf16 Kl[2][KBLK * DHEAD];   // [32 keys][64] x2
  __shared__ __align__(16) bf16 Vl[2][DHEAD * KBLK];   // [64 d][32 keys] x2
  __shared__ __align__(16) bf16 Plds[4][16 * 32];      // per-wave P staging

  int wid  = threadIdx.x >> 5;
  int lane = threadIdx.x & 31;
  int half = lane >> 4, cn = lane & 15;
  int bh    = blockIdx.y;
  int qbase = blockIdx.x * 64;
  int qlo   = qbase + wid * 16;
  const bf16* Qb = Q  + ((size_t)bh * TSEQ + qlo) * DHEAD;
  const bf16* Kb = K  + (size_t)bh * TSEQ * DHEAD;
  const bf16* Vb = Vt + (size_t)bh * DHEAD * TSEQ;
  bf16* Pw = &Plds[wid][0];

  bf16x16 qa0 = ld_tile(Qb, DHEAD);
  bf16x16 qa1 = ld_tile(Qb + 32, DHEAD);

  f32x8 a0 = {}, a1 = {}, a2 = {}, a3 = {};
  float mrow[8], lrow[8];
#pragma unroll
  for (int i = 0; i < 8; ++i) { mrow[i] = -3.0e38f; lrow[i] = 0.0f; }

  const float L2E = 1.4426950408889634f;
  int nkb = blockIdx.x * 2 + 2;                        // shared causal bound

  if (wid == 0) {                                      // preload block 0
    tdm_load_2d(Kb, &Kl[0][0], DHEAD, TSEQ, DHEAD, KBLK, DHEAD);
    tdm_load_2d(Vb, &Vl[0][0], TSEQ, DHEAD, KBLK, DHEAD, TSEQ);
  }

  for (int kb = 0; kb < nkb; ++kb) {
    int kt0 = kb * KBLK;
    int cur = kb & 1;
    if (wid == 0) {
      if (kb + 1 < nkb) {                              // prefetch next block
        int nt0 = kt0 + KBLK;
        tdm_load_2d(Kb + (size_t)nt0 * DHEAD, &Kl[cur ^ 1][0],
                    DHEAD, TSEQ, DHEAD, KBLK, DHEAD);
        tdm_load_2d(Vb + nt0, &Vl[cur ^ 1][0],
                    TSEQ, DHEAD, KBLK, DHEAD, TSEQ);
        wait_tensorcnt_2();                            // current block done
      } else {
        wait_tensorcnt_0();
      }
    }
    __syncthreads();                                   // publish current tiles

    if (kt0 <= qlo + 15) {                             // wave-uniform causal skip
      const bf16* Kt = &Kl[cur][0];
      const bf16* Vv = &Vl[cur][0];
      // ---- S = Q*K^T (scale pre-folded into Q) ----
      f32x8 s0 = {}, s1 = {};
      bf16x16 t0 = ld_tile(Kt, DHEAD);
      bf16x16 t1 = ld_tile(Kt + 32, DHEAD);
      s0 = wmma_bf16(qa0, t0, s0);
      s0 = wmma_bf16(qa1, t1, s0);
      bf16x16 t2 = ld_tile(Kt + 16 * DHEAD, DHEAD);
      bf16x16 t3 = ld_tile(Kt + 16 * DHEAD + 32, DHEAD);
      s1 = wmma_bf16(qa0, t2, s1);
      s1 = wmma_bf16(qa1, t3, s1);
      // ---- online softmax (DPP reductions, raw v_exp) ----
#pragma unroll
      for (int i = 0; i < 8; ++i) {
        int qrow = qlo + i + 8 * half;
        float x0 = s0[i];
        float x1 = s1[i];
        if (kt0 + cn > qrow)      x0 = -3.0e38f;
        if (kt0 + 16 + cn > qrow) x1 = -3.0e38f;
        float nm   = fmaxf(mrow[i], red_max16(fmaxf(x0, x1)));
        float corr = fast_exp2((mrow[i] - nm) * L2E);
        mrow[i] = nm;
        float p0 = fast_exp2((x0 - nm) * L2E);
        float p1 = fast_exp2((x1 - nm) * L2E);
        lrow[i] = lrow[i] * corr + red_sum16(p0 + p1);
        a0[i] *= corr; a1[i] *= corr; a2[i] *= corr; a3[i] *= corr;
        int prow = i + 8 * half;
        Pw[prow * 32 + cn]      = (bf16)p0;
        Pw[prow * 32 + 16 + cn] = (bf16)p1;
      }
      // ---- acc += P * V ----
      bf16x16 pa = ld_tile(Pw, 32);                    // same-wave ds ordering
      a0 = wmma_bf16(pa, ld_tile(Vv + 0 * 16 * KBLK, KBLK), a0);
      a1 = wmma_bf16(pa, ld_tile(Vv + 1 * 16 * KBLK, KBLK), a1);
      a2 = wmma_bf16(pa, ld_tile(Vv + 2 * 16 * KBLK, KBLK), a2);
      a3 = wmma_bf16(pa, ld_tile(Vv + 3 * 16 * KBLK, KBLK), a3);
    }
    __syncthreads();                                   // protect buffer reuse
  }

  // ---- normalize, write attn output as [B, T, H*64] bf16 ----
  int b = bh >> 4, h = bh & 15;
#pragma unroll
  for (int i = 0; i < 8; ++i) {
    float inv = lrow[i] > 0.0f ? 1.0f / lrow[i] : 0.0f;
    int t = qlo + i + 8 * half;
    bf16* orow = O + ((size_t)b * TSEQ + t) * DMODEL + h * DHEAD;
    orow[0 * 16 + cn] = (bf16)(a0[i] * inv);
    orow[1 * 16 + cn] = (bf16)(a1[i] * inv);
    orow[2 * 16 + cn] = (bf16)(a2[i] * inv);
    orow[3 * 16 + cn] = (bf16)(a3[i] * inv);
  }
}

// ---------------------------------------------------------------------------
extern "C" void kernel_launch(void* const* d_in, const int* in_sizes, int n_in,
                              void* d_out, int out_size, void* d_ws, size_t ws_size,
                              hipStream_t stream) {
  (void)in_sizes; (void)n_in; (void)out_size; (void)ws_size;
  const float* q  = (const float*)d_in[0];
  const float* k  = (const float*)d_in[1];
  const float* v  = (const float*)d_in[2];
  // d_in[3] = causal mask (recomputed analytically in-kernel)
  const float* Wq = (const float*)d_in[4];
  const float* Wk = (const float*)d_in[5];
  const float* Wv = (const float*)d_in[6];
  const float* Wo = (const float*)d_in[7];

  const size_t WSZ = (size_t)DMODEL * DMODEL;   // 1M elems
  const size_t XSZ = (size_t)MTOK * DMODEL;     // 8M elems
  bf16* wqb = (bf16*)d_ws;
  bf16* wkb = wqb + WSZ;
  bf16* wvb = wkb + WSZ;
  bf16* wob = wvb + WSZ;
  bf16* Xb  = wob + WSZ;
  bf16* Qb  = Xb + XSZ;
  bf16* Kb  = Qb + XSZ;
  bf16* Vtb = Kb + XSZ;
  bf16* AOb = Xb;                               // attn out reuses X buffer

  dim3 cb(256);
  dim3 cw((unsigned)((WSZ / 4 + 255) / 256));
  dim3 cx((unsigned)((XSZ / 4 + 255) / 256));
  cvt_f32_bf16<<<cw, cb, 0, stream>>>(Wq, wqb, (int)(WSZ / 4), 0.125f); // fold 1/sqrt(64)
  cvt_f32_bf16<<<cw, cb, 0, stream>>>(Wk, wkb, (int)(WSZ / 4), 1.0f);
  cvt_f32_bf16<<<cw, cb, 0, stream>>>(Wv, wvb, (int)(WSZ / 4), 1.0f);
  cvt_f32_bf16<<<cw, cb, 0, stream>>>(Wo, wob, (int)(WSZ / 4), 1.0f);

  dim3 gg(MTOK / 64, DMODEL / 64);
  dim3 gb(128);
  cvt_f32_bf16<<<cx, cb, 0, stream>>>(q, Xb, (int)(XSZ / 4), 1.0f);
  gemm_nt_kernel<0><<<gg, gb, 0, stream>>>(Xb, wqb, Qb);
  cvt_f32_bf16<<<cx, cb, 0, stream>>>(k, Xb, (int)(XSZ / 4), 1.0f);
  gemm_nt_kernel<0><<<gg, gb, 0, stream>>>(Xb, wkb, Kb);
  cvt_f32_bf16<<<cx, cb, 0, stream>>>(v, Xb, (int)(XSZ / 4), 1.0f);
  gemm_nt_kernel<1><<<gg, gb, 0, stream>>>(Xb, wvb, Vtb);

  flash_attn_kernel<<<dim3(TSEQ / 64, BATCH * NHEADS), gb, 0, stream>>>(Qb, Kb, Vtb, AOb);

  gemm_nt_kernel<2><<<gg, gb, 0, stream>>>(AOb, wob, d_out);
}
